// GINEConvZINC_77008763617629
// MI455X (gfx1250) — compile-verified, matrix operand order checked by reading, not run on hardware
//
#include <hip/hip_runtime.h>
#include <hip/hip_bf16.h>
#include <hip/hip_fp16.h>
#include <cstddef>
#include <cstdint>

// ---------------- problem constants ----------------
#define HDIM 128
#define ADIM 64
#define PEDIM 16
#define NLAYERS 4
#define BN_EPS 1e-5f

typedef _Float16 v16h __attribute__((ext_vector_type(16)));
typedef _Float16 v8h  __attribute__((ext_vector_type(8)));
typedef float    v8f  __attribute__((ext_vector_type(8)));
typedef int      v4i  __attribute__((ext_vector_type(4)));

// gfx1250 async global->LDS copy path (ASYNCcnt), if toolchain exposes it
#if __has_builtin(__builtin_amdgcn_global_load_async_to_lds_b128) && __has_builtin(__builtin_amdgcn_s_wait_asynccnt)
#define GINE_ASYNC_LDS 1
typedef __attribute__((address_space(1))) v4i* gptr_v4i;   // global
typedef __attribute__((address_space(3))) v4i* lptr_v4i;   // LDS
#endif

// ---------------- small prep kernels ----------------

// eval-BN folded into scale/shift:  y = x*scale + shift
__global__ void gine_bn_prep(const float* __restrict__ g, const float* __restrict__ b,
                             const float* __restrict__ m, const float* __restrict__ v,
                             float* __restrict__ scale, float* __restrict__ shift, int n) {
    int i = blockIdx.x * blockDim.x + threadIdx.x;
    if (i < n) {
        float s = g[i] * rsqrtf(v[i] + BN_EPS);
        scale[i] = s;
        shift[i] = b[i] - m[i] * s;
    }
}

// Re-shuffle a row-major f32 weight matrix [K][128] into the exact per-lane
// B-fragment layout of v_wmma_f32_16x16x32_f16 (wave32), converted to f16.
// dst index = ((kt*8 + nt)*512 + lane*16 + half)
__global__ void gine_frag_weights(const float* __restrict__ src, _Float16* __restrict__ dst, int K) {
    int idx = blockIdx.x * blockDim.x + threadIdx.x;
    int total = (K >> 5) * 8 * 512;
    if (idx >= total) return;
    int half = idx & 15;
    int lane = (idx >> 4) & 31;
    int nt   = (idx >> 9) & 7;
    int kt   = idx >> 12;
    int ksel = lane >> 4;
    int n    = nt * 16 + (lane & 15);
    int koff = (half < 8) ? (ksel * 8 + half) : (16 + ksel * 8 + (half - 8));
    int k    = kt * 32 + koff;
    dst[idx] = (_Float16)src[(size_t)k * HDIM + n];
}

// Per-node: p = BN(ReLU(pe@peW1+b1)@peW2+b2); concat = [atom_emb[x], p] as f16.
// One 128-thread block per node.
__global__ void gine_pe_concat(const float* __restrict__ pe,
                               const float* __restrict__ peW1, const float* __restrict__ peb1,
                               const float* __restrict__ peW2, const float* __restrict__ peb2,
                               const float* __restrict__ pesc, const float* __restrict__ pesh,
                               const float* __restrict__ atom_emb, const int* __restrict__ x_idx,
                               _Float16* __restrict__ concat, int N) {
    __shared__ float hid[HDIM];
    int node = blockIdx.x;
    if (node >= N) return;
    int t = threadIdx.x;
    float s = peb1[t];
    const float* per = pe + (size_t)node * PEDIM;
#pragma unroll
    for (int j = 0; j < PEDIM; ++j) s += per[j] * peW1[j * HDIM + t];
    hid[t] = fmaxf(s, 0.0f);
    __syncthreads();
    if (t < ADIM) {
        float p = peb2[t];
#pragma unroll 8
        for (int k = 0; k < HDIM; ++k) p += hid[k] * peW2[k * ADIM + t];
        p = p * pesc[t] + pesh[t];
        concat[(size_t)node * HDIM + ADIM + t] = (_Float16)p;
    } else {
        int c = t - ADIM;
        concat[(size_t)node * HDIM + c] = (_Float16)atom_emb[(size_t)x_idx[node] * ADIM + c];
    }
}

// ---------------- WMMA GEMM: Y[rows,128] = f16( act( ((X + addin) @ W + bias)*scale+shift ) ) ----
// X: f16 [rows,K] row-major; addin: optional f16 [rows,K] (aggregated messages);
// Wfrag: pre-fragmentized f16. 128 threads = 4 waves; each wave does a 16x128 tile.
__global__ __launch_bounds__(128) void gine_gemm_wmma(
    const _Float16* __restrict__ X, const _Float16* __restrict__ addin,
    const _Float16* __restrict__ Wfrag, const float* __restrict__ bias,
    const float* __restrict__ scale, const float* __restrict__ shift,
    int do_relu, int nrows, int K, _Float16* __restrict__ Y) {
    // padded row stride (halves) so row-strided ds_load_b128 avoids bank conflicts
    __shared__ __align__(16) _Float16 xs[64 * 264];
    const int stride = K + 8;
    const int row0 = blockIdx.x * 64;
    const int t = threadIdx.x;

    // ---- stage 64 rows into LDS (b128 chunks of 8 halves) ----
    const int CH   = K >> 3;                 // chunks per row (16 or 32)
    const int chsh = (CH == 16) ? 4 : 5;
    const int nchunks = 64 * CH;
#if defined(GINE_ASYNC_LDS)
    if (!addin) {
        // pure f16 copy: async global->LDS, no VGPR round trip (ASYNCcnt path)
        for (int c = t; c < nchunks; c += 128) {
            int r = c >> chsh, kc = c & (CH - 1);
            int row = row0 + r;
            int rr = (row < nrows) ? row : (nrows - 1);
            __builtin_amdgcn_global_load_async_to_lds_b128(
                (gptr_v4i)(X + (size_t)rr * K + kc * 8),
                (lptr_v4i)(&xs[r * stride + kc * 8]),
                0, 0);
        }
        __builtin_amdgcn_s_wait_asynccnt(0);
    } else
#endif
    {
        for (int c = t; c < nchunks; c += 128) {
            int r = c >> chsh, kc = c & (CH - 1);
            int row = row0 + r;
            int rr = (row < nrows) ? row : (nrows - 1);
            v8h xv = *((const v8h*)(X + (size_t)rr * K) + kc);
            if (addin) xv = xv + *((const v8h*)(addin + (size_t)rr * K) + kc);
            *(v8h*)(&xs[r * stride + kc * 8]) = xv;
        }
    }
    __syncthreads();

    const int wave = t >> 5;
    const int lane = t & 31;
    const int m16  = lane & 15;
    const int ksel = lane >> 4;
    const int rbase = wave * 16;

    v8f acc[8] = {};

    const int ktn = K >> 5;
    for (int kt = 0; kt < ktn; ++kt) {
        int k0 = kt * 32;
        // A fragment per ISA 16-bit 16x32 layout (two 16B LDS loads)
        const _Float16* pa = &xs[(rbase + m16) * stride + k0 + ksel * 8];
        v8h lo = *(const v8h*)pa;
        v8h hi = *(const v8h*)(pa + 16);
        v16h a;
#pragma unroll
        for (int i = 0; i < 8; ++i) { a[i] = lo[i]; a[8 + i] = hi[i]; }
#pragma unroll
        for (int nt = 0; nt < 8; ++nt) {
            const _Float16* pb = Wfrag + ((size_t)(kt * 8 + nt) * 512 + lane * 16);
            v16h b = *(const v16h*)pb;   // contiguous 32B per lane
            acc[nt] = __builtin_amdgcn_wmma_f32_16x16x32_f16(
                false, a, false, b, (short)0, acc[nt], false, false);
        }
    }

    // fused epilogue: bias + BN(scale,shift) + optional ReLU, store f16
#pragma unroll
    for (int nt = 0; nt < 8; ++nt) {
        int n = nt * 16 + m16;
        float bi = bias  ? bias[n]  : 0.0f;
        float sc = scale ? scale[n] : 1.0f;
        float sh = shift ? shift[n] : 0.0f;
#pragma unroll
        for (int r = 0; r < 8; ++r) {
            int row = row0 + rbase + r + (ksel ? 8 : 0);
            if (row < nrows) {
                float v = (acc[nt][r] + bi) * sc + sh;
                if (do_relu) v = fmaxf(v, 0.0f);
                Y[(size_t)row * HDIM + n] = (_Float16)v;
            }
        }
    }
}

// ---------------- message passing / pooling ----------------

__global__ void gine_zero_f32(float* __restrict__ p, size_t n) {
    size_t i = (size_t)blockIdx.x * blockDim.x + threadIdx.x;
    if (i < n) p[i] = 0.0f;
}

// msg = relu(h[src] + bond_emb[attr]); agg[dst] += msg.
// 16 threads per edge, 8 channels each; packed-half atomics (global_atomic_pk_add_f16).
__global__ void gine_message(const _Float16* __restrict__ h, const float* __restrict__ bond_emb,
                             const int* __restrict__ ei, const int* __restrict__ eattr,
                             _Float16* __restrict__ agg, int E) {
    long tid = (long)blockIdx.x * blockDim.x + threadIdx.x;
    long e = tid >> 4;
    int c8 = (int)(tid & 15) * 8;
    if (e >= E) return;
    int src = ei[e];
    int dst = ei[(size_t)E + e];
    const float* eb = bond_emb + (size_t)eattr[e] * HDIM + c8;
    v8h hv = *(const v8h*)(h + (size_t)src * HDIM + c8);
    __half2* ap = (__half2*)(agg + (size_t)dst * HDIM + c8);
#pragma unroll
    for (int i = 0; i < 4; ++i) {
        float m0 = fmaxf((float)hv[2 * i]     + eb[2 * i],     0.0f);
        float m1 = fmaxf((float)hv[2 * i + 1] + eb[2 * i + 1], 0.0f);
        atomicAdd(ap + i, __floats2half2_rn(m0, m1));
    }
}

__global__ void gine_pool_add(const _Float16* __restrict__ h, const int* __restrict__ batch,
                              float* __restrict__ padd, int N) {
    long tid = (long)blockIdx.x * blockDim.x + threadIdx.x;
    if (tid >= (long)N * HDIM) return;
    int node = (int)(tid >> 7);
    int c = (int)(tid & 127);
    atomicAdd(padd + (size_t)batch[node] * HDIM + c, (float)h[tid]);
}

__global__ void gine_pool_cnt(const int* __restrict__ batch, float* __restrict__ cnt, int N) {
    int i = blockIdx.x * blockDim.x + threadIdx.x;
    if (i < N) atomicAdd(cnt + batch[i], 1.0f);
}

// gmat[g] = f16 concat([add, add/max(cnt,1)])  -> [G, 256]
__global__ void gine_gmat(const float* __restrict__ padd, const float* __restrict__ cnt,
                          _Float16* __restrict__ gmat, int G) {
    long tid = (long)blockIdx.x * blockDim.x + threadIdx.x;
    if (tid >= (long)G * 256) return;
    int g = (int)(tid >> 8);
    int c = (int)(tid & 255);
    float v;
    if (c < HDIM) v = padd[(size_t)g * HDIM + c];
    else          v = padd[(size_t)g * HDIM + (c - HDIM)] / fmaxf(cnt[g], 1.0f);
    gmat[tid] = (_Float16)v;
}

// out[g] = dot(y[g,:], rW2) + rb2  (y already ReLU'd by GEMM epilogue)
__global__ void gine_readout_final(const _Float16* __restrict__ y, const float* __restrict__ rW2,
                                   const float* __restrict__ rb2, float* __restrict__ out, int G) {
    __shared__ float red[HDIM];
    int g = blockIdx.x;
    if (g >= G) return;
    int t = threadIdx.x;
    red[t] = (float)y[(size_t)g * HDIM + t] * rW2[t];
    __syncthreads();
    for (int s = 64; s > 0; s >>= 1) {
        if (t < s) red[t] += red[t + s];
        __syncthreads();
    }
    if (t == 0) out[g] = red[0] + rb2[0];
}

// ---------------- host-side orchestration ----------------

extern "C" void kernel_launch(void* const* d_in, const int* in_sizes, int n_in,
                              void* d_out, int out_size, void* d_ws, size_t ws_size,
                              hipStream_t stream) {
    const int* x_idx      = (const int*)  d_in[0];
    const int* edge_index = (const int*)  d_in[1];
    const int* edge_attr  = (const int*)  d_in[2];
    const int* batch      = (const int*)  d_in[3];
    const float* pe       = (const float*)d_in[4];
    const float* atom_emb = (const float*)d_in[5];
    const float* bond_emb = (const float*)d_in[6];
    const float* peW1     = (const float*)d_in[7];
    const float* peb1     = (const float*)d_in[8];
    const float* peW2     = (const float*)d_in[9];
    const float* peb2     = (const float*)d_in[10];
    const float* pbg      = (const float*)d_in[11];
    const float* pbb      = (const float*)d_in[12];
    const float* pbm      = (const float*)d_in[13];
    const float* pbv      = (const float*)d_in[14];
    const float* inW      = (const float*)d_in[15];
    const float* inb      = (const float*)d_in[16];
    const float* convW1   = (const float*)d_in[17];
    const float* convb1   = (const float*)d_in[18];
    const float* bn1g     = (const float*)d_in[19];
    const float* bn1b     = (const float*)d_in[20];
    const float* bn1m     = (const float*)d_in[21];
    const float* bn1v     = (const float*)d_in[22];
    const float* convW2   = (const float*)d_in[23];
    const float* convb2   = (const float*)d_in[24];
    const float* bng      = (const float*)d_in[25];
    const float* bnb      = (const float*)d_in[26];
    const float* bnm      = (const float*)d_in[27];
    const float* bnv      = (const float*)d_in[28];
    const float* rW1      = (const float*)d_in[29];
    const float* rb1      = (const float*)d_in[30];
    const float* rW2      = (const float*)d_in[31];
    const float* rb2      = (const float*)d_in[32];

    const int N = in_sizes[0];
    const int E = in_sizes[2];
    const int G = out_size;
    float* out = (float*)d_out;

    // ---- workspace carving (256B aligned) ----
    char* base = (char*)d_ws;
    size_t off = 0;
    auto carve = [&](size_t bytes) -> char* {
        char* p = base + off;
        off += (bytes + 255) & ~(size_t)255;
        return p;
    };
    _Float16* h16   = (_Float16*)carve((size_t)N * HDIM * 2);
    _Float16* t16   = (_Float16*)carve((size_t)N * HDIM * 2);   // also holds concat
    _Float16* agg   = (_Float16*)carve((size_t)N * HDIM * 2);   // f16 message accumulator
    _Float16* fIn   = (_Float16*)carve(16384 * 2);
    _Float16* fW1   = (_Float16*)carve((size_t)NLAYERS * 16384 * 2);
    _Float16* fW2   = (_Float16*)carve((size_t)NLAYERS * 16384 * 2);
    _Float16* fR1   = (_Float16*)carve(32768 * 2);               // K=256
    float*    pesc  = (float*)   carve(ADIM * 4);
    float*    pesh  = (float*)   carve(ADIM * 4);
    float*    s1tab = (float*)   carve((size_t)NLAYERS * HDIM * 4);
    float*    t1tab = (float*)   carve((size_t)NLAYERS * HDIM * 4);
    float*    s2tab = (float*)   carve((size_t)NLAYERS * HDIM * 4);
    float*    t2tab = (float*)   carve((size_t)NLAYERS * HDIM * 4);
    float*    padd  = (float*)   carve((size_t)G * HDIM * 4);
    float*    cnt   = (float*)   carve((size_t)G * 4);
    _Float16* gmat  = (_Float16*)carve((size_t)G * 256 * 2);
    _Float16* y16   = (_Float16*)carve((size_t)G * HDIM * 2);
    (void)ws_size; (void)n_in;

    // ---- BN fold ----
    gine_bn_prep<<<1, ADIM, 0, stream>>>(pbg, pbb, pbm, pbv, pesc, pesh, ADIM);
    gine_bn_prep<<<(NLAYERS * HDIM + 127) / 128, 128, 0, stream>>>(bn1g, bn1b, bn1m, bn1v, s1tab, t1tab, NLAYERS * HDIM);
    gine_bn_prep<<<(NLAYERS * HDIM + 127) / 128, 128, 0, stream>>>(bng, bnb, bnm, bnv, s2tab, t2tab, NLAYERS * HDIM);

    // ---- fragmentize all weights once (f32 -> WMMA-layout f16) ----
    gine_frag_weights<<<64, 256, 0, stream>>>(inW, fIn, HDIM);
    for (int l = 0; l < NLAYERS; ++l) {
        gine_frag_weights<<<64, 256, 0, stream>>>(convW1 + (size_t)l * HDIM * HDIM, fW1 + (size_t)l * 16384, HDIM);
        gine_frag_weights<<<64, 256, 0, stream>>>(convW2 + (size_t)l * HDIM * HDIM, fW2 + (size_t)l * 16384, HDIM);
    }
    gine_frag_weights<<<128, 256, 0, stream>>>(rW1, fR1, 2 * HDIM);

    // ---- node features: concat([atom_emb gather, BN(pe MLP)]) -> f16 ----
    gine_pe_concat<<<N, HDIM, 0, stream>>>(pe, peW1, peb1, peW2, peb2, pesc, pesh,
                                           atom_emb, x_idx, t16, N);

    // ---- input projection: h = concat @ inW + inb ----
    int gblk = (N + 63) / 64;
    gine_gemm_wmma<<<gblk, 128, 0, stream>>>(t16, nullptr, fIn, inb,
                                             nullptr, nullptr, 0, N, HDIM, h16);

    // ---- GINE layers ----
    size_t nh = (size_t)N * HDIM;
    int zblk = (int)((nh / 2 + 255) / 256);                    // agg is f16: nh/2 f32 words
    int mblk = (int)(((long)E * 16 + 255) / 256);
    for (int l = 0; l < NLAYERS; ++l) {
        gine_zero_f32<<<zblk, 256, 0, stream>>>((float*)agg, nh / 2);
        gine_message<<<mblk, 256, 0, stream>>>(h16, bond_emb, edge_index, edge_attr, agg, E);
        // z = (h+agg)@W1+b1 -> BN -> ReLU
        gine_gemm_wmma<<<gblk, 128, 0, stream>>>(h16, agg, fW1 + (size_t)l * 16384,
                                                 convb1 + (size_t)l * HDIM,
                                                 s1tab + (size_t)l * HDIM, t1tab + (size_t)l * HDIM,
                                                 1, N, HDIM, t16);
        // h = ReLU(BN(z@W2+b2))
        gine_gemm_wmma<<<gblk, 128, 0, stream>>>(t16, nullptr, fW2 + (size_t)l * 16384,
                                                 convb2 + (size_t)l * HDIM,
                                                 s2tab + (size_t)l * HDIM, t2tab + (size_t)l * HDIM,
                                                 1, N, HDIM, h16);
    }

    // ---- pooling ----
    size_t gh = (size_t)G * HDIM;
    gine_zero_f32<<<(int)((gh + 255) / 256), 256, 0, stream>>>(padd, gh);
    gine_zero_f32<<<(G + 255) / 256, 256, 0, stream>>>(cnt, (size_t)G);
    gine_pool_add<<<(int)((nh + 255) / 256), 256, 0, stream>>>(h16, batch, padd, N);
    gine_pool_cnt<<<(N + 255) / 256, 256, 0, stream>>>(batch, cnt, N);
    gine_gmat<<<(int)(((long)G * 256 + 255) / 256), 256, 0, stream>>>(padd, cnt, gmat, G);

    // ---- readout: ReLU(gmat @ rW1 + rb1) via WMMA (K=256), then dot with rW2 ----
    gine_gemm_wmma<<<(G + 63) / 64, 128, 0, stream>>>(gmat, nullptr, fR1, rb1,
                                                      nullptr, nullptr, 1, G, 2 * HDIM, y16);
    gine_readout_final<<<G, HDIM, 0, stream>>>(y16, rW2, rb2, out, G);
}
